// LocalTransformerNet_68513318306013
// MI455X (gfx1250) — matrix operand order repaired and checked by reading, not compiled, not run on hardware
//
#include <hip/hip_runtime.h>

// ---------------------------------------------------------------------------
// LocalTransformerNet for MI455X (gfx1250, wave32).
// Dense token GEMMs: v_wmma_f32_16x16x32_f16 (f16 A/B, f32 accum).
// Weights pre-packed into WMMA-B lane layout (1 aligned v16h load per tile).
// Windowed attention (25 tok x hd=8 x 4 heads): VALU + LDS + f32 atomics.
// ---------------------------------------------------------------------------

typedef __attribute__((ext_vector_type(16))) _Float16 v16h;
typedef __attribute__((ext_vector_type(8)))  _Float16 v8h;
typedef __attribute__((ext_vector_type(8)))  float    v8f;
typedef __attribute__((ext_vector_type(4)))  float    v4f;

#define NB    4
#define HPDIM 256
#define LSZ   512
#define NTOK  (NB * HPDIM * HPDIM)   // 262144 tokens
#define MT    (NTOK / 16)            // 16384 M-tiles
#define WPB   8                      // waves per 256-thread block
#define GEMM_BLOCKS (MT / WPB)       // 2048

// Packed weight tile ids (each tile: 32 lanes x 16 f16 = 1024 B)
#define PATCH_T 0   // 2 tiles
#define QKV_T   2   // 6 tiles
#define PROJ_T  8   // 2 tiles
#define MLP1_T  10  // 4 tiles
#define MLP2_T  14  // 4 tiles: 14=(k0,n0) 15=(k0,n1) 16=(k1,n0) 17=(k1,n1)
#define OUT_T   18  // 3 tiles
#define NTILES  21

// ---------------- WMMA helpers (layouts per CDNA5 ISA) ----------

static __device__ __forceinline__ v8f wmma_f16(v16h a, v16h b, v8f c) {
  return __builtin_amdgcn_wmma_f32_16x16x32_f16(false, a, false, b, (short)0, c,
                                                false, false);
}

// 16-bit A 16x32: lane g=0 holds K=0..7 (v0..3) and 16..23 (v4..7);
// lane g=1 holds K=8..15 and 24..31. m = lane&15.
static __device__ __forceinline__ v16h load_a_f16(const _Float16* __restrict__ src,
                                                  int tok0, int ld, int m, int g,
                                                  int kbase) {
  const _Float16* row = src + (size_t)(tok0 + m) * ld + kbase;
  v8h lo = *(const v8h*)(row + 8 * g);        // 16B aligned
  v8h hi = *(const v8h*)(row + 16 + 8 * g);   // 16B aligned
  return __builtin_shufflevector(lo, hi, 0, 1, 2, 3, 4, 5, 6, 7,
                                 8, 9, 10, 11, 12, 13, 14, 15);
}

static __device__ __forceinline__ v16h load_a_f32(const float* __restrict__ src,
                                                  int tok0, int ld, int m, int g,
                                                  int kbase) {
  const float* row = src + (size_t)(tok0 + m) * ld + kbase;
  v4f x0 = *(const v4f*)(row + 8 * g);
  v4f x1 = *(const v4f*)(row + 8 * g + 4);
  v4f x2 = *(const v4f*)(row + 16 + 8 * g);
  v4f x3 = *(const v4f*)(row + 16 + 8 * g + 4);
  v16h a;
#pragma unroll
  for (int e = 0; e < 4; ++e) {
    a[e]      = (_Float16)x0[e];
    a[4 + e]  = (_Float16)x1[e];
    a[8 + e]  = (_Float16)x2[e];
    a[12 + e] = (_Float16)x3[e];
  }
  return a;
}

// Pre-packed B tile: one aligned 32-byte load per lane.
static __device__ __forceinline__ v16h load_bp(const _Float16* __restrict__ wpack,
                                               int tile, int lane) {
  return *(const v16h*)(wpack + (size_t)tile * 512 + lane * 16);
}

// C/D 16x16 f32: VGPR i <-> row (i + 8*g), column = lane&15 (+ ncol0).

// ---------------- Weight pre-pack into WMMA-B lane layout ------------------
// B 32x16 f16: lanes 0-15 hold K=0..7,16..23, lanes 16-31 hold K=8..15,24..31; n = lane&15.

__global__ __launch_bounds__(32) void k_pack_w(
    const float* __restrict__ pw, const float* __restrict__ qw,
    const float* __restrict__ prw, const float* __restrict__ w1,
    const float* __restrict__ w2, const float* __restrict__ ow,
    _Float16* __restrict__ wpack) {
  int tile = blockIdx.x;
  int lane = threadIdx.x;
  int n = lane & 15, g = lane >> 4;
  const float* W; int Kreal, Ntot, kbase, ncol0;
  if (tile < QKV_T)       { W = pw;  Kreal = 24; Ntot = 32; kbase = 0; ncol0 = (tile - PATCH_T) * 16; }
  else if (tile < PROJ_T) { W = qw;  Kreal = 32; Ntot = 96; kbase = 0; ncol0 = (tile - QKV_T) * 16; }
  else if (tile < MLP1_T) { W = prw; Kreal = 32; Ntot = 32; kbase = 0; ncol0 = (tile - PROJ_T) * 16; }
  else if (tile < MLP2_T) { W = w1;  Kreal = 32; Ntot = 64; kbase = 0; ncol0 = (tile - MLP1_T) * 16; }
  else if (tile < OUT_T)  { W = w2;  Kreal = 64; Ntot = 32; kbase = ((tile - MLP2_T) >> 1) * 32;
                            ncol0 = ((tile - MLP2_T) & 1) * 16; }
  else                    { W = ow;  Kreal = 32; Ntot = 48; kbase = 0; ncol0 = (tile - OUT_T) * 16; }
  v16h b;
#pragma unroll
  for (int e = 0; e < 16; ++e) {
    int k = kbase + 8 * g + (e < 8 ? e : e + 8);
    b[e] = (k < Kreal) ? (_Float16)W[(size_t)k * Ntot + ncol0 + n] : (_Float16)0.f;
  }
  *(v16h*)(wpack + (size_t)tile * 512 + lane * 16) = b;
}

// ---------------- Kernel 1: patchify + patch embed (24->32) ----------------
// A gather per lane: g=0 needs channels {0,1} (K0..7) and {4,5} (K16..23);
// g=1 needs channels {2,3} (K8..15); K24..31 are zero pad.
// Each channel contributes a 2x2 pixel patch = two float2 loads.

static __device__ __forceinline__ const float* chan_plane(
    const float* __restrict__ plaq, const float* __restrict__ rect,
    int b, int c) {
  const float* p0 = plaq + ((size_t)b * 2 + c) * (size_t)LSZ * LSZ;
  const float* p1 = rect + ((size_t)b * 4 + (c - 2)) * (size_t)LSZ * LSZ;
  return (c < 2) ? p0 : p1;   // pointer select, no exec branch
}

__global__ __launch_bounds__(256) void k_patch_embed(
    const float* __restrict__ plaq, const float* __restrict__ rect,
    const _Float16* __restrict__ wpack, const float* __restrict__ pb,
    float* __restrict__ patches) {
  int lane = threadIdx.x & 31, wave = threadIdx.x >> 5;
  int tok0 = (blockIdx.x * WPB + wave) * 16;
  int m = lane & 15, g = lane >> 4;
  int tok = tok0 + m;
  int b  = tok >> 16;
  int hp = (tok >> 8) & 255;
  int wp = tok & 255;
  size_t pix0 = (size_t)(2 * hp) * LSZ + 2 * wp;   // 8B aligned (2*wp even)

  int   c0 = g ? 2 : 0;       // channels for elements 0..7
  int   c1 = g ? 2 : 4;       // channels for elements 8..15 (dummy when g=1)
  float mk = g ? 0.f : 1.f;   // zero mask for K>=24 pad

  v16h a;
#pragma unroll
  for (int ch = 0; ch < 2; ++ch) {   // elements 0..7
    const float* P = chan_plane(plaq, rect, b, c0 + ch);
    float2 r0 = *(const float2*)(P + pix0);
    float2 r1 = *(const float2*)(P + pix0 + LSZ);
    a[ch * 4 + 0] = (_Float16)r0.x;
    a[ch * 4 + 1] = (_Float16)r0.y;
    a[ch * 4 + 2] = (_Float16)r1.x;
    a[ch * 4 + 3] = (_Float16)r1.y;
  }
#pragma unroll
  for (int ch = 0; ch < 2; ++ch) {   // elements 8..15 (masked to 0 when g=1)
    const float* P = chan_plane(plaq, rect, b, c1 + ch);
    float2 r0 = *(const float2*)(P + pix0);
    float2 r1 = *(const float2*)(P + pix0 + LSZ);
    a[8 + ch * 4 + 0] = (_Float16)(r0.x * mk);
    a[8 + ch * 4 + 1] = (_Float16)(r0.y * mk);
    a[8 + ch * 4 + 2] = (_Float16)(r1.x * mk);
    a[8 + ch * 4 + 3] = (_Float16)(r1.y * mk);
  }
#pragma unroll
  for (int nt = 0; nt < 2; ++nt) {
    int ncol0 = nt * 16;
    v16h bm = load_bp(wpack, PATCH_T + nt, lane);
    v8f c;
#pragma unroll
    for (int i = 0; i < 8; ++i) c[i] = pb[ncol0 + m];
    c = wmma_f16(a, bm, c);
#pragma unroll
    for (int i = 0; i < 8; ++i)
      patches[(size_t)(tok0 + i + 8 * g) * 32 + ncol0 + m] = c[i];
  }
}

// ---------------- LayerNorm over 32 features, emit f16 ---------------------

__global__ __launch_bounds__(256) void k_layernorm(
    const float* __restrict__ x, const float* __restrict__ gw,
    const float* __restrict__ gb, _Float16* __restrict__ out) {
  int tok = blockIdx.x * blockDim.x + threadIdx.x;
  if (tok >= NTOK) return;
  const float* row = x + (size_t)tok * 32;
  float r[32];
#pragma unroll
  for (int j = 0; j < 8; ++j) {
    v4f v = *(const v4f*)(row + j * 4);
#pragma unroll
    for (int e = 0; e < 4; ++e) r[j * 4 + e] = v[e];
  }
  float s = 0.f;
#pragma unroll
  for (int j = 0; j < 32; ++j) s += r[j];
  float mean = s * (1.f / 32.f);
  float v = 0.f;
#pragma unroll
  for (int j = 0; j < 32; ++j) { float d = r[j] - mean; v += d * d; }
  float rstd = rsqrtf(v * (1.f / 32.f) + 1e-5f);
  _Float16* o = out + (size_t)tok * 32;
#pragma unroll
  for (int j = 0; j < 4; ++j) {
    v8h h;
#pragma unroll
    for (int e = 0; e < 8; ++e) {
      int jj = j * 8 + e;
      h[e] = (_Float16)((r[jj] - mean) * rstd * gw[jj] + gb[jj]);
    }
    *(v8h*)(o + j * 8) = h;
  }
}

// ---------------- QKV GEMM 32 -> 96 (f16 out) ------------------------------

__global__ __launch_bounds__(256) void k_qkv(
    const _Float16* __restrict__ xln, const _Float16* __restrict__ wpack,
    const float* __restrict__ bias, _Float16* __restrict__ qkv) {
  int lane = threadIdx.x & 31, wave = threadIdx.x >> 5;
  int tok0 = (blockIdx.x * WPB + wave) * 16;
  int m = lane & 15, g = lane >> 4;
  v16h a = load_a_f16(xln, tok0, 32, m, g, 0);
#pragma unroll
  for (int nt = 0; nt < 6; ++nt) {
    int ncol0 = nt * 16;
    v16h bm = load_bp(wpack, QKV_T + nt, lane);
    v8f c;
#pragma unroll
    for (int i = 0; i < 8; ++i) c[i] = bias[ncol0 + m];
    c = wmma_f16(a, bm, c);
#pragma unroll
    for (int i = 0; i < 8; ++i)
      qkv[(size_t)(tok0 + i + 8 * g) * 96 + ncol0 + m] = (_Float16)c[i];
  }
}

// ---------------- zero fill ------------------------------------------------

__global__ __launch_bounds__(256) void k_zero(float* __restrict__ p, long n4) {
  long i = (long)blockIdx.x * blockDim.x + threadIdx.x;
  if (i < n4) *( (v4f*)p + i ) = (v4f){0.f, 0.f, 0.f, 0.f};
}

// ---------------- Windowed attention: 1 wave per 5x5 window ----------------
// Overlapping windows (stride 2) scatter-add into acc[B,Hp,Hp,32] (f32 atomics).

__global__ __launch_bounds__(128) void k_attn(
    const _Float16* __restrict__ qkv, float* __restrict__ acc) {
  __shared__ float S[4][25 * 96];
  int lane = threadIdx.x & 31, wave = threadIdx.x >> 5;
  int wid = blockIdx.x * 4 + wave;             // 65536 windows total
  int b  = wid >> 14;
  int rm = wid & 16383;
  int mw = rm >> 7;                            // window row 0..127
  int nw = rm & 127;                           // window col 0..127
  float* W = S[wave];
  int t = lane;
  if (t < 25) {                                // stage window qkv into LDS (f32)
    int r  = 2 * mw + t / 5;
    int cc = 2 * nw + t % 5;
    int rc = min(r, 255), ccc = min(cc, 255);  // clamped gather (ref semantics)
    size_t tok = ((size_t)b * 256 + rc) * 256 + ccc;
    const v8h* src = (const v8h*)(qkv + tok * 96);   // 192B row, 16B aligned
    float* dst = W + t * 96;
#pragma unroll
    for (int i = 0; i < 12; ++i) {
      v8h h8 = src[i];
      float4 f0 = {(float)h8[0], (float)h8[1], (float)h8[2], (float)h8[3]};
      float4 f1 = {(float)h8[4], (float)h8[5], (float)h8[6], (float)h8[7]};
      *(float4*)(dst + i * 8)     = f0;
      *(float4*)(dst + i * 8 + 4) = f1;
    }
  }
  __syncthreads();
  if (t >= 25) return;
  int r  = 2 * mw + t / 5;
  int cc = 2 * nw + t % 5;
  bool qvalid = (r < 256) && (cc < 256);
  const float scale = 0.35355339059327373f;    // 1/sqrt(hd=8)
#pragma unroll
  for (int h = 0; h < 4; ++h) {
    float4 qa = *(const float4*)&W[t * 96 + h * 8];
    float4 qb = *(const float4*)&W[t * 96 + h * 8 + 4];
    float s[25];
    float mx = -1e30f;
#pragma unroll
    for (int kk = 0; kk < 25; ++kk) {
      int kr = 2 * mw + kk / 5, kc = 2 * nw + kk % 5;
      bool kv = (kr < 256) && (kc < 256);
      float4 ka = *(const float4*)&W[kk * 96 + 32 + h * 8];
      float4 kb = *(const float4*)&W[kk * 96 + 32 + h * 8 + 4];
      float dot = qa.x * ka.x + qa.y * ka.y + qa.z * ka.z + qa.w * ka.w +
                  qb.x * kb.x + qb.y * kb.y + qb.z * kb.z + qb.w * kb.w;
      s[kk] = kv ? dot * scale : -1e9f;
      mx = fmaxf(mx, s[kk]);
    }
    float sum = 0.f;
#pragma unroll
    for (int kk = 0; kk < 25; ++kk) { s[kk] = __expf(s[kk] - mx); sum += s[kk]; }
    float inv = 1.f / sum;
    float4 oa = {0.f, 0.f, 0.f, 0.f};
    float4 ob = {0.f, 0.f, 0.f, 0.f};
#pragma unroll
    for (int kk = 0; kk < 25; ++kk) {
      float p = s[kk] * inv;
      float4 va = *(const float4*)&W[kk * 96 + 64 + h * 8];
      float4 vb = *(const float4*)&W[kk * 96 + 64 + h * 8 + 4];
      oa.x += p * va.x; oa.y += p * va.y; oa.z += p * va.z; oa.w += p * va.w;
      ob.x += p * vb.x; ob.y += p * vb.y; ob.z += p * vb.z; ob.w += p * vb.w;
    }
    if (qvalid) {
      float* dst = acc + (((size_t)b * 256 + r) * 256 + cc) * 32 + h * 8;
      atomicAdd(&dst[0], oa.x); atomicAdd(&dst[1], oa.y);
      atomicAdd(&dst[2], oa.z); atomicAdd(&dst[3], oa.w);
      atomicAdd(&dst[4], ob.x); atomicAdd(&dst[5], ob.y);
      atomicAdd(&dst[6], ob.z); atomicAdd(&dst[7], ob.w);
    }
  }
}

// ---------------- proj 32->32 (+bias +residual via C operand) --------------

__global__ __launch_bounds__(256) void k_proj(
    const float* __restrict__ accf, const _Float16* __restrict__ wpack,
    const float* __restrict__ bias, const float* __restrict__ res,
    float* __restrict__ out) {
  int lane = threadIdx.x & 31, wave = threadIdx.x >> 5;
  int tok0 = (blockIdx.x * WPB + wave) * 16;
  int m = lane & 15, g = lane >> 4;
  v16h a = load_a_f32(accf, tok0, 32, m, g, 0);
#pragma unroll
  for (int nt = 0; nt < 2; ++nt) {
    int ncol0 = nt * 16;
    v16h bm = load_bp(wpack, PROJ_T + nt, lane);
    v8f c;
#pragma unroll
    for (int i = 0; i < 8; ++i)
      c[i] = bias[ncol0 + m] + res[(size_t)(tok0 + i + 8 * g) * 32 + ncol0 + m];
    c = wmma_f16(a, bm, c);
#pragma unroll
    for (int i = 0; i < 8; ++i)
      out[(size_t)(tok0 + i + 8 * g) * 32 + ncol0 + m] = c[i];
  }
}

// ---------------- MLP1 32->64 + exact GELU (f16 out) -----------------------

__global__ __launch_bounds__(256) void k_mlp1(
    const _Float16* __restrict__ hln, const _Float16* __restrict__ wpack,
    const float* __restrict__ b1, _Float16* __restrict__ out) {
  int lane = threadIdx.x & 31, wave = threadIdx.x >> 5;
  int tok0 = (blockIdx.x * WPB + wave) * 16;
  int m = lane & 15, g = lane >> 4;
  v16h a = load_a_f16(hln, tok0, 32, m, g, 0);
#pragma unroll
  for (int nt = 0; nt < 4; ++nt) {
    int ncol0 = nt * 16;
    v16h bm = load_bp(wpack, MLP1_T + nt, lane);
    v8f c;
#pragma unroll
    for (int i = 0; i < 8; ++i) c[i] = b1[ncol0 + m];
    c = wmma_f16(a, bm, c);
#pragma unroll
    for (int i = 0; i < 8; ++i) {
      float x = c[i];
      float ge = 0.5f * x * (1.f + erff(x * 0.70710678118654752f));
      out[(size_t)(tok0 + i + 8 * g) * 64 + ncol0 + m] = (_Float16)ge;
    }
  }
}

// ---------------- MLP2 64->32 (K=64: two WMMA accumulations) ---------------

__global__ __launch_bounds__(256) void k_mlp2(
    const _Float16* __restrict__ m1, const _Float16* __restrict__ wpack,
    const float* __restrict__ b2, const float* __restrict__ res,
    float* __restrict__ out) {
  int lane = threadIdx.x & 31, wave = threadIdx.x >> 5;
  int tok0 = (blockIdx.x * WPB + wave) * 16;
  int m = lane & 15, g = lane >> 4;
  v16h a0 = load_a_f16(m1, tok0, 64, m, g, 0);
  v16h a1 = load_a_f16(m1, tok0, 64, m, g, 32);
#pragma unroll
  for (int nt = 0; nt < 2; ++nt) {
    int ncol0 = nt * 16;
    v8f c;
#pragma unroll
    for (int i = 0; i < 8; ++i)
      c[i] = b2[ncol0 + m] + res[(size_t)(tok0 + i + 8 * g) * 32 + ncol0 + m];
    c = wmma_f16(a0, load_bp(wpack, MLP2_T + nt, lane), c);      // K=0..31
    c = wmma_f16(a1, load_bp(wpack, MLP2_T + 2 + nt, lane), c);  // K=32..63
#pragma unroll
    for (int i = 0; i < 8; ++i)
      out[(size_t)(tok0 + i + 8 * g) * 32 + ncol0 + m] = c[i];
  }
}

// ---------------- out 32->48 + tanh*0.25 + un-patchify scatter -------------

__global__ __launch_bounds__(256) void k_out(
    const float* __restrict__ p3, const _Float16* __restrict__ wpack,
    const float* __restrict__ bias, float* __restrict__ dout) {
  int lane = threadIdx.x & 31, wave = threadIdx.x >> 5;
  int tok0 = (blockIdx.x * WPB + wave) * 16;
  int m = lane & 15, g = lane >> 4;
  v16h a = load_a_f32(p3, tok0, 32, m, g, 0);
#pragma unroll
  for (int nt = 0; nt < 3; ++nt) {
    int ncol0 = nt * 16;
    v16h bm = load_bp(wpack, OUT_T + nt, lane);
    v8f c;
#pragma unroll
    for (int i = 0; i < 8; ++i) c[i] = bias[ncol0 + m];
    c = wmma_f16(a, bm, c);
    int f  = ncol0 + m;          // f = c_out*4 + dy*2 + dx
    int co = f >> 2;
    int dy = (f >> 1) & 1;
    int dx = f & 1;
#pragma unroll
    for (int i = 0; i < 8; ++i) {
      int tok = tok0 + i + 8 * g;
      int b  = tok >> 16;
      int hp = (tok >> 8) & 255;
      int wp = tok & 255;
      float val = tanhf(c[i]) * 0.25f;
      size_t row = (size_t)(2 * hp + dy);
      size_t col = (size_t)(2 * wp + dx);
      size_t off = (co < 4)
        ? ((((size_t)b * 4 + co) * LSZ + row) * LSZ + col)
        : ((size_t)NB * 4 * LSZ * LSZ +
           (((size_t)b * 8 + (co - 4)) * LSZ + row) * LSZ + col);
      dout[off] = val;
    }
  }
}

// ---------------- host-side pipeline ---------------------------------------

extern "C" void kernel_launch(void* const* d_in, const int* in_sizes, int n_in,
                              void* d_out, int out_size, void* d_ws, size_t ws_size,
                              hipStream_t stream) {
  (void)in_sizes; (void)n_in; (void)out_size; (void)ws_size;
  const float* plaq    = (const float*)d_in[0];
  const float* rect    = (const float*)d_in[1];
  const float* patch_w = (const float*)d_in[2];
  const float* patch_b = (const float*)d_in[3];
  const float* n1g     = (const float*)d_in[4];
  const float* n1b     = (const float*)d_in[5];
  const float* qkv_w   = (const float*)d_in[6];
  const float* qkv_b   = (const float*)d_in[7];
  const float* proj_w  = (const float*)d_in[8];
  const float* proj_b  = (const float*)d_in[9];
  const float* n2g     = (const float*)d_in[10];
  const float* n2b     = (const float*)d_in[11];
  const float* w1      = (const float*)d_in[12];
  const float* b1      = (const float*)d_in[13];
  const float* w2      = (const float*)d_in[14];
  const float* b2      = (const float*)d_in[15];
  const float* ow      = (const float*)d_in[16];
  const float* ob      = (const float*)d_in[17];

  char* ws = (char*)d_ws;
  float*    patches  = (float*)(ws + 0);                 // 33.5 MB f32
  _Float16* xln      = (_Float16*)(ws + 33554432ull);    // 16.8 MB f16
  _Float16* qkvb     = (_Float16*)(ws + 50331648ull);    // 50.3 MB f16
  float*    accb     = (float*)(ws + 100663296ull);      // 33.5 MB f32
  float*    patches2 = (float*)(ws + 134217728ull);      // 33.5 MB f32
  _Float16* wpack    = (_Float16*)(ws + 167772160ull);   // 21 KB packed B tiles
  _Float16* hln      = xln;       // xln dead after k_qkv
  _Float16* mlp1b    = qkvb;      // qkv dead after k_attn (needs 33.5 MB)
  float*    patches3 = patches;   // patches dead after k_proj
  float*    dout     = (float*)d_out;

  dim3 blk(256);
  k_pack_w<<<NTILES, dim3(32), 0, stream>>>(patch_w, qkv_w, proj_w, w1, w2, ow, wpack);
  k_patch_embed<<<GEMM_BLOCKS, blk, 0, stream>>>(plaq, rect, wpack, patch_b, patches);
  k_layernorm<<<NTOK / 256, blk, 0, stream>>>(patches, n1g, n1b, xln);
  k_qkv<<<GEMM_BLOCKS, blk, 0, stream>>>(xln, wpack, qkv_b, qkvb);
  k_zero<<<(NTOK * 32 / 4) / 256, blk, 0, stream>>>(accb, (long)NTOK * 32 / 4);
  k_attn<<<(NB * 128 * 128) / 4, dim3(128), 0, stream>>>(qkvb, accb);
  k_proj<<<GEMM_BLOCKS, blk, 0, stream>>>(accb, wpack, proj_b, patches, patches2);
  k_layernorm<<<NTOK / 256, blk, 0, stream>>>(patches2, n2g, n2b, hln);
  k_mlp1<<<GEMM_BLOCKS, blk, 0, stream>>>(hln, wpack, b1, mlp1b);
  k_mlp2<<<GEMM_BLOCKS, blk, 0, stream>>>(mlp1b, wpack, b2, patches2, patches3);
  k_out<<<GEMM_BLOCKS, blk, 0, stream>>>(patches3, wpack, ob, dout);
}
